// MODEL_41996190220781
// MI455X (gfx1250) — matrix operand order repaired and checked by reading, not compiled, not run on hardware
//
#include <hip/hip_runtime.h>
#include <hip/hip_bf16.h>

#define ALPHA   0.2f
#define NEG_INF (-9.0e15f)
#define MT      4                      // 16-row sub-tiles per block (M = 64)

typedef __attribute__((ext_vector_type(16))) __bf16 v16bf;
typedef __attribute__((ext_vector_type(8)))  __bf16 v8bf;
typedef __attribute__((ext_vector_type(8)))  float  v8f;

// ---- WMMA fragment loaders (CDNA5 16-bit layouts, wave32) ----------------
// A 16x32 bf16: lane m=lane&15; lanes>=16 shifted by K+8. VGPR0..3 = K+0..7,
// VGPR4..7 = K+16..23  -> two 16B loads, 32B apart.
__device__ __forceinline__ v16bf load_frag_a(const __bf16* p) {
    v8bf lo = *(const v8bf*)(p);
    v8bf hi = *(const v8bf*)(p + 16);
    v16bf r;
#pragma unroll
    for (int i = 0; i < 8; ++i) { r[i] = lo[i]; r[i + 8] = hi[i]; }
    return r;
}
// B 32x16 bf16: lane n=lane&15; lanes>=16 shifted by K+8. VGPR0..3 = K+0..7,
// VGPR4..7 = K+16..23 (same as A) -> two 16B loads, 32B apart.
__device__ __forceinline__ v16bf load_frag_b(const __bf16* p) {
    v8bf lo = *(const v8bf*)(p);
    v8bf hi = *(const v8bf*)(p + 16);
    v16bf r;
#pragma unroll
    for (int i = 0; i < 8; ++i) { r[i] = lo[i]; r[i + 8] = hi[i]; }
    return r;
}

// ---- Phase 1: kc_Wh = kc_h @ W1 (store transposed bf16), kc_score --------
__global__ void k_kc(const float* __restrict__ kc_h, const float* __restrict__ W1,
                     const float* __restrict__ a, __bf16* __restrict__ kcT,
                     float* __restrict__ kc_score) {
    __shared__ float kch[256];
    __shared__ float red[256];
    const int j = blockIdx.x;          // kc row
    const int t = threadIdx.x;         // output column
    kch[t] = kc_h[j * 256 + t];
    __syncthreads();
    float s = 0.f;
#pragma unroll 8
    for (int k = 0; k < 256; ++k) s = fmaf(kch[k], W1[k * 256 + t], s);
    kcT[(size_t)t * 1024 + j] = (__bf16)s;          // kc_Wh^T, K-contiguous
    red[t] = s * a[256 + t];
    __syncthreads();
    for (int off = 128; off > 0; off >>= 1) {
        if (t < off) red[t] += red[t + off];
        __syncthreads();
    }
    if (t == 0) kc_score[j] = red[0];
}

// ---- Phase 2: E^T bf16 and w1a = W1 @ a[:d] ------------------------------
__global__ void k_prep(const float* __restrict__ W1, const float* __restrict__ E,
                       const float* __restrict__ a, __bf16* __restrict__ ET,
                       float* __restrict__ w1a) {
    const int t = threadIdx.x;
    if (blockIdx.x < 256) {
        const int n = blockIdx.x;
        ET[(size_t)n * 256 + t] = (__bf16)E[(size_t)t * 256 + n];
    } else {
        __shared__ float av[256];
        av[t] = a[t];
        __syncthreads();
        float s = 0.f;
#pragma unroll 8
        for (int c = 0; c < 256; ++c) s = fmaf(W1[(size_t)t * 256 + c], av[c], s);
        w1a[t] = s;
    }
}

// ---- Phase 3: ex_score = exercise_h @ w1a (wave-per-row GEMV) ------------
__global__ void k_exscore(const float* __restrict__ ex_h, const float* __restrict__ w1a,
                          float* __restrict__ ex_score, int n_rows) {
    __shared__ float wv[256];
    const int t = threadIdx.x;
    wv[t] = w1a[t];
    __syncthreads();
    const int wave = t >> 5, lane = t & 31;
    const int row = blockIdx.x * 8 + wave;
    if (row >= n_rows) return;
    float s = 0.f;
#pragma unroll
    for (int j = 0; j < 8; ++j) {
        const int k = lane + 32 * j;
        s = fmaf(ex_h[(size_t)row * 256 + k], wv[k], s);
    }
#pragma unroll
    for (int off = 16; off > 0; off >>= 1) s += __shfl_down(s, off, 32);
    if (lane == 0) ex_score[row] = s;
}

// ---- Phase 4: fused mask + softmax + (attn@kcWh) * (h@E) -> elu ----------
// 64 exercise rows per block (4x16 sub-tiles) so each B fragment feeds 4
// WMMAs from registers -> 4x less L2 traffic on kc_Wh^T.
__global__ void __launch_bounds__(256)
k_main(const float* __restrict__ ex_h, const int* __restrict__ adj,
       const float* __restrict__ ex_score, const float* __restrict__ kc_score,
       const __bf16* __restrict__ kcT, const __bf16* __restrict__ ET,
       float* __restrict__ out, int n_tiles16) {
    __shared__ float  kcs[1024];            //   4 KB
    __shared__ __bf16 probs[MT * 16][1024]; // 128 KB unnormalized numerators
    __shared__ __bf16 exh[MT * 16][256];    //  32 KB
    __shared__ float  red[16][16];          //   1 KB (reused per chunk)
    __shared__ float  invs[MT * 16];
    __shared__ float  exsc[MT * 16];

    const int tid   = threadIdx.x;
    const int tile0 = blockIdx.x * MT;            // first 16-row tile index
    const int nt    = min(MT, n_tiles16 - tile0); // valid sub-tiles (uniform)
    const int m0    = tile0 * 16;
    const int nrows = nt * 16;

    for (int i = tid; i < 1024; i += 256) kcs[i] = kc_score[i];
    if (tid < nrows) exsc[tid] = ex_score[m0 + tid];
    for (int i = tid; i < nrows * 256; i += 256) {
        const int r = i >> 8, c = i & 255;
        exh[r][c] = (__bf16)ex_h[(size_t)(m0 + r) * 256 + c];
    }
    __syncthreads();

    // --- masked leaky-relu logits + softmax, one 16-row chunk at a time ---
    const int row = tid >> 4, sub = tid & 15;
    for (int c = 0; c < nt; ++c) {
        const int   lr = c * 16 + row;            // local row
        const float es = exsc[lr];
        float lv[64];
        float mx = -3.0e38f;
#pragma unroll 4
        for (int j = 0; j < 64; ++j) {
            const int col = sub + (j << 4);
            float s = es + kcs[col];
            s = (s > 0.f) ? s : ALPHA * s;
            s = (adj[(size_t)(m0 + lr) * 1024 + col] > 0) ? s : NEG_INF;
            lv[j] = s;
            mx = fmaxf(mx, s);
        }
        red[row][sub] = mx;
        __syncthreads();
        float rmax = red[row][0];
#pragma unroll
        for (int i = 1; i < 16; ++i) rmax = fmaxf(rmax, red[row][i]);

        float ps = 0.f;
#pragma unroll 4
        for (int j = 0; j < 64; ++j) {
            const float p = __expf(lv[j] - rmax);  // masked -> 0; row max -> 1
            ps += p;
            probs[lr][sub + (j << 4)] = (__bf16)p;
        }
        __syncthreads();               // rmax reads done before red reuse
        red[row][sub] = ps;
        __syncthreads();
        if (sub == 0) {
            float s = 0.f;
#pragma unroll
            for (int i = 0; i < 16; ++i) s += red[row][i];
            invs[lr] = 1.0f / s;       // s >= 1 (max-subtracted), never 0
        }
        __syncthreads();
    }

    // --- per-wave WMMA: 2 N-tiles per wave, B fragment reused over 4 M ---
    const int wv_  = tid >> 5, lane = tid & 31;
    const int hiHf = lane >> 4;                // 0: lanes 0-15, 1: lanes 16-31
    const int mA   = lane & 15;
    const int kA   = hiHf ? 8 : 0;             // A-frag half-wave K shift
    const int nB   = lane & 15;
    const int kB   = hiHf ? 8 : 0;             // B-frag half-wave K shift

#pragma unroll
    for (int t = 0; t < 2; ++t) {
        const int n0 = (2 * wv_ + t) * 16;
        const __bf16* bcol = kcT + (size_t)(n0 + nB) * 1024;

        v8f accA[MT] = {};
#pragma unroll 2
        for (int kt = 0; kt < 32; ++kt) {
            const int kb = kt * 32;
            __builtin_prefetch(bcol + kb + 32, 0, 1);      // global_prefetch_b8
            v16bf bf = load_frag_b(bcol + kb + kB);
#pragma unroll
            for (int c = 0; c < MT; ++c) {
                v16bf af = load_frag_a(&probs[c * 16 + mA][kb + kA]);
                accA[c] = __builtin_amdgcn_wmma_f32_16x16x32_bf16(
                    false, af, false, bf, (short)0, accA[c], false, false);
            }
        }

        const __bf16* ecol = ET + (size_t)(n0 + nB) * 256;
        v8f accE[MT] = {};
#pragma unroll
        for (int kt = 0; kt < 8; ++kt) {
            const int kb = kt * 32;
            v16bf bf = load_frag_b(ecol + kb + kB);
#pragma unroll
            for (int c = 0; c < MT; ++c) {
                v16bf af = load_frag_a(&exh[c * 16 + mA][kb + kA]);
                accE[c] = __builtin_amdgcn_wmma_f32_16x16x32_bf16(
                    false, af, false, bf, (short)0, accE[c], false, false);
            }
        }

        // C/D layout: VGPR r -> M = r + 8*hiHalf, N = lane&15
        for (int c = 0; c < nt; ++c) {
#pragma unroll
            for (int r = 0; r < 8; ++r) {
                const int m = c * 16 + r + (hiHf ? 8 : 0);
                float v = accA[c][r] * invs[m] * accE[c][r];
                v = (v > 0.f) ? v : (__expf(v) - 1.0f);    // elu
                out[(size_t)(m0 + m) * 256 + n0 + nB] = v;
            }
        }
    }
}

// ---- Host launcher --------------------------------------------------------
extern "C" void kernel_launch(void* const* d_in, const int* in_sizes, int n_in,
                              void* d_out, int out_size, void* d_ws, size_t ws_size,
                              hipStream_t stream) {
    const float* ex_h = (const float*)d_in[0];   // [N_ex, 256]
    const float* kc_h = (const float*)d_in[1];   // [1024, 256]
    const int*   adj  = (const int*)  d_in[2];   // [N_ex, 1024]
    const float* W1   = (const float*)d_in[3];   // [256, 256]
    const float* E    = (const float*)d_in[4];   // [256, 256]
    const float* a    = (const float*)d_in[5];   // [512, 1]
    float* out = (float*)d_out;

    const int N_ex     = in_sizes[0] / 256;      // 50000
    const int n_tiles16 = N_ex / 16;             // 3125 (exact)

    char* w = (char*)d_ws;
    __bf16* kcT      = (__bf16*)w; w += (size_t)1024 * 256 * sizeof(__bf16); // kc_Wh^T
    __bf16* ET       = (__bf16*)w; w += (size_t)256 * 256 * sizeof(__bf16);  // E^T
    float*  kc_score = (float*)w;  w += 1024 * sizeof(float);
    float*  w1a      = (float*)w;  w += 256 * sizeof(float);
    float*  ex_score = (float*)w;  w += (size_t)N_ex * sizeof(float);

    k_kc<<<1024, 256, 0, stream>>>(kc_h, W1, a, kcT, kc_score);
    k_prep<<<257, 256, 0, stream>>>(W1, E, a, ET, w1a);
    k_exscore<<<(N_ex + 7) / 8, 256, 0, stream>>>(ex_h, w1a, ex_score, N_ex);
    k_main<<<(n_tiles16 + MT - 1) / MT, 256, 0, stream>>>(
        ex_h, adj, ex_score, kc_score, kcT, ET, out, n_tiles16);
}
